// SimpleSetTopoLayer_70317204570673
// MI455X (gfx1250) — matrix operand order repaired and checked by reading, not compiled, not run on hardware
//
#include <hip/hip_runtime.h>

typedef _Float16 v16h __attribute__((ext_vector_type(16)));
typedef _Float16 v8h  __attribute__((ext_vector_type(8)));
typedef float    v8f  __attribute__((ext_vector_type(8)));

#define NG   2048
#define F    128
#define H    128
#define IN0  144
#define K1P  160   // layer-1 K padded to multiple of 32
#define AST1 168   // LDS stride (halves) layer 1 (bank-conflict free)
#define AST2 136   // LDS stride (halves) layer 2
#define ROWS_SS 512

__device__ inline v16h hcombine(v8h lo, v8h hi) {
  v16h r;
#pragma unroll
  for (int i = 0; i < 8; ++i) { r[i] = lo[i]; r[i + 8] = hi[i]; }
  return r;
}

__global__ void k_zero(float* p, long n) {
  long i = (long)blockIdx.x * blockDim.x + threadIdx.x;
  long st = (long)gridDim.x * blockDim.x;
  for (; i < n; i += st) p[i] = 0.f;
}

// weights -> f16 (layer1 K padded 144->160 with zeros)
__global__ void k_cvtw(const float* g1w, const float* g2w,
                       _Float16* g1wh, _Float16* g2wh) {
  int t = blockIdx.x * 256 + threadIdx.x;
  if (t < 128 * K1P) {
    int col = t / K1P, k = t - col * K1P;
    g1wh[t] = (_Float16)(k < IN0 ? g1w[col * IN0 + k] : 0.f);
  }
  if (t < 128 * 128) g2wh[t] = (_Float16)g2w[t];
}

// segment sums of x0 = [x | pers0^T]; batch is sorted -> run-length accumulate,
// one atomic per (segment transition, feature). Thread f==144 counts rows.
__global__ void k_segsum0(const float* __restrict__ x, const float* __restrict__ pers0,
                          const int* __restrict__ batch,
                          float* segsum, float* cnt, int N) {
  int f = threadIdx.x;                 // blockDim = 160; active f <= 144
  long r0 = (long)blockIdx.x * ROWS_SS;
  if (r0 >= N || f > IN0) return;
  long r1 = r0 + ROWS_SS; if (r1 > N) r1 = N;
  const bool isCnt = (f == IN0);
  int cur = batch[r0];
  float acc = 0.f;
  for (long row = r0; row < r1; ++row) {
    int sg = batch[row];
    if (sg != cur) {
      if (isCnt) atomicAdd(&cnt[cur], acc);
      else       atomicAdd(&segsum[cur * IN0 + f], acc);
      cur = sg; acc = 0.f;
    }
    float v;
    if (isCnt) v = 1.f;
    else if (f < F) v = x[row * F + f];
    else { int q = f - F; v = pers0[(long)(q >> 1) * 2 * N + row * 2 + (q & 1)]; }
    acc += v;
  }
  if (isCnt) atomicAdd(&cnt[cur], acc);
  else       atomicAdd(&segsum[cur * IN0 + f], acc);
}

// segment sums of out1 (f16)
__global__ void k_segsum1(const _Float16* __restrict__ out1h, const int* __restrict__ batch,
                          float* segsum, int N) {
  int f = threadIdx.x;                 // blockDim = 128
  long r0 = (long)blockIdx.x * ROWS_SS;
  if (r0 >= N) return;
  long r1 = r0 + ROWS_SS; if (r1 > N) r1 = N;
  int cur = batch[r0];
  float acc = 0.f;
  for (long row = r0; row < r1; ++row) {
    int sg = batch[row];
    if (sg != cur) { atomicAdd(&segsum[cur * H + f], acc); cur = sg; acc = 0.f; }
    acc += (float)out1h[row * H + f];
  }
  atomicAdd(&segsum[cur * H + f], acc);
}

// m[g][h] = (segsum[g]/max(cnt,1)) . lw[h]   (tiny: 2048 x 128 x K)
__global__ void k_proj(const float* __restrict__ segsum, const float* __restrict__ cnt,
                       const float* __restrict__ lw, float* m, int K) {
  int g = blockIdx.x, h = threadIdx.x;   // 128 threads
  float inv = 1.f / fmaxf(cnt[g], 1.f);
  float acc = 0.f;
  for (int k = 0; k < K; ++k) acc += segsum[g * K + k] * lw[h * K + k];
  m[g * 128 + h] = acc * inv;
}

// GEMM-1: out1 = relu(x0 . g1w^T + g1b - m1[seg])  -> f16
// Software-pipelined: next A tile gathered into registers during WMMA compute.
__global__ __launch_bounds__(256) void k_gemm1(
    const float* __restrict__ x, const float* __restrict__ pers0,
    const int* __restrict__ batch, const _Float16* __restrict__ g1wh,
    const float* __restrict__ g1b, const float* __restrict__ m1,
    _Float16* __restrict__ out1h, int N) {
  __shared__ _Float16 Bsm[128 * AST1];
  __shared__ _Float16 Asm[16 * AST1];
  __shared__ int Seg[16];
  const int tid = threadIdx.x;
  const int wave = tid >> 5, lane = tid & 31;
  const int hs = lane >> 4, ln = lane & 15;
  const int col = wave * 16 + ln;

  for (int e = tid; e < 128 * (K1P / 8); e += 256) {
    int c0 = e / (K1P / 8), s8 = e - c0 * (K1P / 8);
    *(v8h*)&Bsm[c0 * AST1 + s8 * 8] = *(const v8h*)&g1wh[c0 * K1P + s8 * 8];
  }
  __syncthreads();

  v16h bf[5];                    // B fragments, invariant over row tiles
#pragma unroll
  for (int ks = 0; ks < 5; ++ks) {
    int k0 = ks * 32 + 16 * hs;
    bf[ks] = hcombine(*(const v8h*)&Bsm[col * AST1 + k0],
                      *(const v8h*)&Bsm[col * AST1 + k0 + 8]);
  }

  const float gb = g1b[col];
  const long base = (long)blockIdx.x * 256;

  float av[10];
  int segReg = 0;
  auto gatherA = [&](long mbase) {
#pragma unroll
    for (int i = 0; i < 10; ++i) {       // 16 rows x 160 (zeros pad, f32 src)
      int e = i * 256 + tid;
      int r = e / K1P, c = e - r * K1P;
      long row = mbase + r;
      float v = 0.f;
      if (row < N && c < IN0) {
        if (c < F) v = x[row * F + c];
        else { int q = c - F; v = pers0[(long)(q >> 1) * 2 * N + row * 2 + (q & 1)]; }
      }
      av[i] = v;
    }
    if (tid < 16) { long row = mbase + tid; segReg = (row < N) ? batch[row] : 0; }
  };

  gatherA(base);                          // tile 0 loads in flight

  for (int t = 0; t < 16; ++t) {
    const long mbase = base + t * 16;
    __syncthreads();                      // all waves done with previous Asm
#pragma unroll
    for (int i = 0; i < 10; ++i) {        // waits on loads, cvt, stage to LDS
      int e = i * 256 + tid;
      int r = e / K1P, c = e - r * K1P;
      Asm[r * AST1 + c] = (_Float16)av[i];
    }
    if (tid < 16) Seg[tid] = segReg;
    __syncthreads();

    if (t < 15) gatherA(mbase + 16);      // next tile loads overlap compute

    // load all A fragments first (ds-load clause), then back-to-back WMMAs
    v16h af[5];
#pragma unroll
    for (int ks = 0; ks < 5; ++ks) {
      int k0 = ks * 32 + 16 * hs;
      af[ks] = hcombine(*(const v8h*)&Asm[ln * AST1 + k0],
                        *(const v8h*)&Asm[ln * AST1 + k0 + 8]);
    }
    v8f acc = {};
#pragma unroll
    for (int ks = 0; ks < 5; ++ks)
      acc = __builtin_amdgcn_wmma_f32_16x16x32_f16(false, af[ks], false, bf[ks],
                                                   (short)0, acc, false, false);

    // epilogue: independent m1 gathers (Seg from LDS), one wait, then stores
    int sg[8];
#pragma unroll
    for (int r = 0; r < 8; ++r) sg[r] = Seg[r + 8 * hs];
    float mv[8];
#pragma unroll
    for (int r = 0; r < 8; ++r) mv[r] = m1[sg[r] * 128 + col];

    const long rb = mbase + 8 * hs;
    if (mbase + 16 <= N) {
#pragma unroll
      for (int r = 0; r < 8; ++r) {
        float v = acc[r] + gb - mv[r];
        v = v > 0.f ? v : 0.f;
        out1h[(rb + r) * H + col] = (_Float16)v;
      }
    } else {
#pragma unroll
      for (int r = 0; r < 8; ++r) {
        if (rb + r < N) {
          float v = acc[r] + gb - mv[r];
          v = v > 0.f ? v : 0.f;
          out1h[(rb + r) * H + col] = (_Float16)v;
        }
      }
    }
  }
}

// GEMM-2: out2 = out1 . g2w^T + g2b - m2[seg]
// mode 0: accumulate BN partial sums (per-lane regs -> LDS -> partials)
// mode 1: final y = x + (out2 - mu) * rstd * gamma + beta
__global__ __launch_bounds__(256) void k_gemm2(
    const _Float16* __restrict__ out1h, const int* __restrict__ batch,
    const _Float16* __restrict__ g2wh, const float* __restrict__ g2b,
    const float* __restrict__ m2, const float* __restrict__ x,
    const float* __restrict__ stats, const float* __restrict__ bng,
    const float* __restrict__ bnb, float* __restrict__ partials,
    float* __restrict__ out, int N, int mode) {
  __shared__ _Float16 Bsm[128 * AST2];
  __shared__ _Float16 Asm[16 * AST2];
  __shared__ int Seg[16];
  __shared__ float ps[256];
  const int tid = threadIdx.x;
  const int wave = tid >> 5, lane = tid & 31;
  const int hs = lane >> 4, ln = lane & 15;
  const int col = wave * 16 + ln;

  ps[tid] = 0.f;

  for (int e = tid; e < 128 * 16; e += 256) {
    int c0 = e >> 4, s8 = e & 15;
    *(v8h*)&Bsm[c0 * AST2 + s8 * 8] = *(const v8h*)&g2wh[c0 * 128 + s8 * 8];
  }
  __syncthreads();

  v16h bf[4];
#pragma unroll
  for (int ks = 0; ks < 4; ++ks) {
    int k0 = ks * 32 + 16 * hs;
    bf[ks] = hcombine(*(const v8h*)&Bsm[col * AST2 + k0],
                      *(const v8h*)&Bsm[col * AST2 + k0 + 8]);
  }

  const float gb = g2b[col];
  float mu = 0.f, rs = 1.f, gam = 1.f, bet = 0.f;
  if (mode == 1) { mu = stats[col]; rs = stats[128 + col]; gam = bng[col]; bet = bnb[col]; }

  float lsum = 0.f, lsq = 0.f;
  const long base = (long)blockIdx.x * 256;
  const int ar = tid >> 4, ac8 = tid & 15;  // A-stage: one v8h per thread

  v8h av = {};
  int segReg = 0;
  auto gatherA = [&](long mbase) {
    long row = mbase + ar;
    v8h v = {};
    if (row < N) v = *(const v8h*)&out1h[row * H + ac8 * 8];
    av = v;
    if (tid < 16) { long r2 = mbase + tid; segReg = (r2 < N) ? batch[r2] : 0; }
  };

  gatherA(base);

  for (int t = 0; t < 16; ++t) {
    const long mbase = base + t * 16;
    __syncthreads();
    *(v8h*)&Asm[ar * AST2 + ac8 * 8] = av;
    if (tid < 16) Seg[tid] = segReg;
    __syncthreads();

    if (t < 15) gatherA(mbase + 16);

    v16h af[4];
#pragma unroll
    for (int ks = 0; ks < 4; ++ks) {
      int k0 = ks * 32 + 16 * hs;
      af[ks] = hcombine(*(const v8h*)&Asm[ln * AST2 + k0],
                        *(const v8h*)&Asm[ln * AST2 + k0 + 8]);
    }
    v8f acc = {};
#pragma unroll
    for (int ks = 0; ks < 4; ++ks)
      acc = __builtin_amdgcn_wmma_f32_16x16x32_f16(false, af[ks], false, bf[ks],
                                                   (short)0, acc, false, false);

    int sg[8];
#pragma unroll
    for (int r = 0; r < 8; ++r) sg[r] = Seg[r + 8 * hs];
    float mv[8];
#pragma unroll
    for (int r = 0; r < 8; ++r) mv[r] = m2[sg[r] * 128 + col];

    const long rb = mbase + 8 * hs;
    const bool full = (mbase + 16 <= N);
    if (mode == 0) {
      if (full) {
#pragma unroll
        for (int r = 0; r < 8; ++r) {
          float v = acc[r] + gb - mv[r];
          lsum += v; lsq += v * v;
        }
      } else {
#pragma unroll
        for (int r = 0; r < 8; ++r) {
          if (rb + r < N) {
            float v = acc[r] + gb - mv[r];
            lsum += v; lsq += v * v;
          }
        }
      }
    } else {
      if (full) {
        float xv[8];
#pragma unroll
        for (int r = 0; r < 8; ++r) xv[r] = x[(rb + r) * F + col];
#pragma unroll
        for (int r = 0; r < 8; ++r) {
          float v = acc[r] + gb - mv[r];
          out[(rb + r) * F + col] = xv[r] + (v - mu) * rs * gam + bet;
        }
      } else {
#pragma unroll
        for (int r = 0; r < 8; ++r) {
          if (rb + r < N) {
            float v = acc[r] + gb - mv[r];
            out[(rb + r) * F + col] = x[(rb + r) * F + col] + (v - mu) * rs * gam + bet;
          }
        }
      }
    }
  }

  if (mode == 0) {
    atomicAdd(&ps[col], lsum);
    atomicAdd(&ps[128 + col], lsq);
    __syncthreads();
    partials[(long)blockIdx.x * 256 + tid] = ps[tid];
  }
}

__global__ void k_reduce(const float* __restrict__ partials, int nblocks,
                         float* stats, int N) {
  __shared__ float s[256];
  int t = threadIdx.x;
  float a = 0.f;
  for (int b = 0; b < nblocks; ++b) a += partials[(long)b * 256 + t];
  s[t] = a;
  __syncthreads();
  if (t < 128) {
    float mu = s[t] / (float)N;
    float var = s[128 + t] / (float)N - mu * mu;
    stats[t] = mu;
    stats[128 + t] = rsqrtf(var + 1e-5f);
  }
}

extern "C" void kernel_launch(void* const* d_in, const int* in_sizes, int n_in,
                              void* d_out, int out_size, void* d_ws, size_t ws_size,
                              hipStream_t stream) {
  (void)n_in; (void)out_size; (void)ws_size;
  const float* x     = (const float*)d_in[0];
  const int*   batch = (const int*)d_in[1];
  const float* pers0 = (const float*)d_in[2];
  const float* g1w   = (const float*)d_in[3];
  const float* g1b   = (const float*)d_in[4];
  const float* l1w   = (const float*)d_in[5];
  const float* g2w   = (const float*)d_in[6];
  const float* g2b   = (const float*)d_in[7];
  const float* l2w   = (const float*)d_in[8];
  const float* bng   = (const float*)d_in[9];
  const float* bnb   = (const float*)d_in[10];
  const int N = in_sizes[0] / F;

  char* ws = (char*)d_ws;
  size_t off = 0;
  auto take = [&](size_t bytes) -> void* {
    void* p = ws + off; off = (off + bytes + 255) & ~(size_t)255; return p;
  };
  float* segsum0   = (float*)take((size_t)NG * IN0 * 4);
  float* cnt       = (float*)take((size_t)NG * 4);
  float* segsum1   = (float*)take((size_t)NG * H * 4);
  long   zeroSpan  = ((char*)(segsum1 + (size_t)NG * H) - (char*)segsum0) / 4;
  float* m1        = (float*)take((size_t)NG * 128 * 4);
  float* m2        = (float*)take((size_t)NG * 128 * 4);
  _Float16* g1wh   = (_Float16*)take((size_t)128 * K1P * 2);
  _Float16* g2wh   = (_Float16*)take((size_t)128 * 128 * 2);
  float* stats     = (float*)take(256 * 4);
  const int nblocks = (N + 255) / 256;
  float* partials  = (float*)take((size_t)nblocks * 256 * 4);
  _Float16* out1h  = (_Float16*)take((size_t)N * H * 2);

  const int ssgrid = (N + ROWS_SS - 1) / ROWS_SS;

  k_zero<<<256, 256, 0, stream>>>(segsum0, zeroSpan);
  k_cvtw<<<(128 * K1P + 255) / 256, 256, 0, stream>>>(g1w, g2w, g1wh, g2wh);
  k_segsum0<<<ssgrid, 160, 0, stream>>>(x, pers0, batch, segsum0, cnt, N);
  k_proj<<<NG, 128, 0, stream>>>(segsum0, cnt, l1w, m1, IN0);
  k_gemm1<<<nblocks, 256, 0, stream>>>(x, pers0, batch, g1wh, g1b, m1, out1h, N);
  k_segsum1<<<ssgrid, 128, 0, stream>>>(out1h, batch, segsum1, N);
  k_proj<<<NG, 128, 0, stream>>>(segsum1, cnt, l2w, m2, H);
  k_gemm2<<<nblocks, 256, 0, stream>>>(out1h, batch, g2wh, g2b, m2, x, stats,
                                       bng, bnb, partials, (float*)d_out, N, 0);
  k_reduce<<<1, 256, 0, stream>>>(partials, nblocks, stats, N);
  k_gemm2<<<nblocks, 256, 0, stream>>>(out1h, batch, g2wh, g2b, m2, x, stats,
                                       bng, bnb, partials, (float*)d_out, N, 1);
}
